// SingleAttention_89215060673176
// MI455X (gfx1250) — matrix operand order repaired and checked by reading, not compiled, not run on hardware
//
#include <hip/hip_runtime.h>
#include <hip/hip_bf16.h>
#include <math.h>

// Problem constants (fixed by the reference)
#define BB 16
#define SS 2048
#define FF 512
#define DD 512

typedef __attribute__((ext_vector_type(16))) __bf16 v16bf;
typedef __attribute__((ext_vector_type(8)))  float  v8f;

typedef unsigned int u32x4 __attribute__((ext_vector_type(4)));
typedef int          i32x8 __attribute__((ext_vector_type(8)));
typedef int          i32x4 __attribute__((ext_vector_type(4)));

#if defined(__gfx1250__) && __has_builtin(__builtin_amdgcn_tensor_load_to_lds)
#define USE_TDM 1
#else
#define USE_TDM 0
#endif

// Native f32 -> bf16 (backend emits packed bf16 cvt instead of bit twiddling)
__device__ __forceinline__ unsigned short f2bf(float f) {
    union { __bf16 h; unsigned short u; } c;
    c.h = (__bf16)f;
    return c.u;
}

// ---------------------------------------------------------------------------
// Kernel 0: W [F,D] fp32  ->  Wt [D,F] bf16  (transposed so WMMA B-fragments
// become two contiguous 16-byte runs per lane)
// ---------------------------------------------------------------------------
__global__ void wt_kernel(const float* __restrict__ W, unsigned short* __restrict__ Wt) {
    int idx = blockIdx.x * blockDim.x + threadIdx.x;   // over D*F
    int d = idx / FF;
    int f = idx - d * FF;
    Wt[idx] = f2bf(W[(long)f * DD + d]);
}

// ---------------------------------------------------------------------------
// Kernel 1: projection  P = X @ W + b   (X fp32 [B*S,F], Wt bf16 [D,F])
// Output bf16: row-major [B*S,D]  (Q,K)  or transposed [B][D][S]  (V)
// Block: 256 thr = 8 waves; block tile 128 rows x 64 cols; wave tile 16x64.
// ---------------------------------------------------------------------------
__global__ __launch_bounds__(256) void proj_kernel(
    const float*          __restrict__ X,
    const unsigned short* __restrict__ Wt,
    const float*          __restrict__ bias,
    unsigned short*       __restrict__ Out,
    int transposed)
{
    const int lane = threadIdx.x & 31;
    const int wave = threadIdx.x >> 5;
    const int half = lane >> 4;        // which 16-lane half
    const int ln   = lane & 15;

    const int row0 = blockIdx.x * 128 + wave * 16;  // wave's 16 output rows
    const int col0 = blockIdx.y * 64;               // wave's 64 output cols

    const int  arow = row0 + ln;       // A-matrix row handled by this lane
    const int  akb  = half * 8;        // A K-base (ISA 16-bit A layout)

    v8f acc[4] = {};                   // four 16x16 f32 C/D fragments

    for (int kt = 0; kt < FF; kt += 32) {
        // ---- A fragment: 16 fp32 -> bf16, per ISA 16x32 layout ----
        union { v16bf v; __bf16 h[16]; } A;
        const float4* a0 = (const float4*)(X + (long)arow * FF + kt + akb);
        const float4* a1 = (const float4*)(X + (long)arow * FF + kt + 16 + akb);
        float4 f0 = a0[0], f1 = a0[1], f2 = a1[0], f3 = a1[1];
        A.h[0]=(__bf16)f0.x; A.h[1]=(__bf16)f0.y; A.h[2]=(__bf16)f0.z; A.h[3]=(__bf16)f0.w;
        A.h[4]=(__bf16)f1.x; A.h[5]=(__bf16)f1.y; A.h[6]=(__bf16)f1.z; A.h[7]=(__bf16)f1.w;
        A.h[8]=(__bf16)f2.x; A.h[9]=(__bf16)f2.y; A.h[10]=(__bf16)f2.z; A.h[11]=(__bf16)f2.w;
        A.h[12]=(__bf16)f3.x; A.h[13]=(__bf16)f3.y; A.h[14]=(__bf16)f3.z; A.h[15]=(__bf16)f3.w;

        #pragma unroll
        for (int t = 0; t < 4; ++t) {
            // ---- B fragment: col n = ln, K = half*8 + 0..7 and 16 + half*8 + 0..7 ----
            int n = col0 + t * 16 + ln;
            const uint4* bq = (const uint4*)(Wt + (long)n * FF + kt + half * 8);
            union { v16bf v; uint4 q[2]; } Bf;
            Bf.q[0] = bq[0]; Bf.q[1] = bq[2];
            acc[t] = __builtin_amdgcn_wmma_f32_16x16x32_bf16(
                false, A.v, false, Bf.v, (short)0, acc[t], false, false);
        }
    }

    if (!transposed) {                 // Q / K: row-major [B*S, D]
        #pragma unroll
        for (int t = 0; t < 4; ++t) {
            int n = col0 + t * 16 + ln;
            float bv = bias[n];
            #pragma unroll
            for (int v = 0; v < 8; ++v) {
                int m = row0 + v + 8 * half;     // C/D layout: row = v + 8*half
                Out[(long)m * DD + n] = f2bf(acc[t][v] + bv);
            }
        }
    } else {                           // V: transposed [B][D][S]
        int b     = row0 / SS;
        int sbase = (row0 % SS) + 8 * half;
        #pragma unroll
        for (int t = 0; t < 4; ++t) {
            int n = col0 + t * 16 + ln;
            float bv = bias[n];
            union { uint4 q; unsigned short us[8]; } pk;
            #pragma unroll
            for (int v = 0; v < 8; ++v) pk.us[v] = f2bf(acc[t][v] + bv);
            *(uint4*)(Out + (long)b * DD * SS + (long)n * SS + sbase) = pk.q;
        }
    }
}

// ---------------------------------------------------------------------------
// Kernel 2: flash attention.
// Block = 256 thr = 8 waves, handles 32 query rows x full D_V for one batch.
// wave = (qw in 0..1)  x  (dw in 0..3)
//   scores: wave (qw,dw) computes 16x16 score frag for keys [ktile+dw*16, +16)
//   output: wave (qw,dw) owns 16 rows x 128 d_v cols (dw*128 slice)
// Key-tile loop of 64 keys with online softmax; P exchanged via LDS as bf16.
// Q tile staged into padded LDS by the Tensor Data Mover (wave 0 issues the
// descriptor; TDM pad feature produces the 520-element row stride).
// ---------------------------------------------------------------------------
#define QSTR 520   // padded LDS row stride (bf16 elems) to spread banks

__global__ __launch_bounds__(256) void attn_kernel(
    const unsigned short* __restrict__ Qb,   // [B*S, D] bf16
    const unsigned short* __restrict__ Kb,   // [B*S, D] bf16
    const unsigned short* __restrict__ Vt,   // [B, D, S] bf16 (transposed)
    float*                __restrict__ Out)  // [B, S, D] f32
{
    __shared__ unsigned short sQ[32 * QSTR];
    __shared__ unsigned short sP[2][16][72];
    __shared__ float sMax[2][4][16];
    __shared__ float sSum[2][4][16];

    const int lane = threadIdx.x & 31;
    const int wave = threadIdx.x >> 5;
    const int qw   = wave >> 2;     // query sub-tile (0..1)
    const int dw   = wave & 3;      // key sub-tile / d_v slice (0..3)
    const int half = lane >> 4;
    const int ln   = lane & 15;

    const int b  = blockIdx.y;
    const int q0 = blockIdx.x * 32;      // query row base within S

#if USE_TDM
    // ---- stage Q tile (32 x 512 bf16) into LDS via Tensor Data Mover ----
    if (wave == 0) {
        unsigned long ga = (unsigned long)(const void*)(Qb + ((long)b * SS + q0) * DD);
        unsigned      la = (unsigned)(unsigned long)(void*)&sQ[0];  // LDS byte offset

        u32x4 g0;
        g0[0] = 1u;                                   // count=1 (valid), user mode
        g0[1] = la;                                   // lds_addr
        g0[2] = (unsigned)(ga & 0xFFFFFFFFu);         // global_addr[31:0]
        g0[3] = (unsigned)((ga >> 32) & 0x1FFFFFFu)   // global_addr[56:32]
              | (2u << 30);                           // type = 2 ("image")

        i32x8 g1;
        g1[0] = (int)((1u << 16)                      // data_size = 2 bytes
              |       (1u << 20)                      // pad_enable
              |       (7u << 22)                      // pad_interval: 256 DWORDs (one row)
              |       (3u << 25));                    // pad_amount: 4 DWORDs (8 bf16)
        g1[1] = (int)(512u << 16);                    // tensor_dim0 = 512 (bits 79:48, low part)
        g1[2] = (int)(32u << 16);                     // tensor_dim0 hi=0 | tensor_dim1=32 (low)
        g1[3] = (int)(512u << 16);                    // tensor_dim1 hi=0 | tile_dim0 = 512
        g1[4] = 32;                                   // tile_dim1 = 32 | tile_dim2 = 0
        g1[5] = 512;                                  // tensor_dim0_stride = 512 (low 32)
        g1[6] = 0;                                    // stride hi | tensor_dim1_stride lo
        g1[7] = 0;

        i32x4 g2 = {0, 0, 0, 0};
        i32x4 g3 = {0, 0, 0, 0};
#if __clang_major__ >= 23
        i32x8 g4 = {0, 0, 0, 0, 0, 0, 0, 0};
        __builtin_amdgcn_tensor_load_to_lds(g0, g1, g2, g3, g4, 0);
#else
        __builtin_amdgcn_tensor_load_to_lds(g0, g1, g2, g3, 0);
#endif
        __builtin_amdgcn_s_wait_tensorcnt(0);
    }
#else
    // ---- fallback: stage Q tile with plain vector loads ----
    {
        int tid = threadIdx.x;
        int r = tid >> 3;                // 0..31
        int c = (tid & 7) * 64;          // 0..448 step 64
        const unsigned short* src = Qb + ((long)b * SS + q0 + r) * DD + c;
        unsigned short*       dst = &sQ[r * QSTR + c];
        #pragma unroll
        for (int i = 0; i < 8; ++i)
            *(uint4*)(dst + i * 8) = *(const uint4*)(src + i * 8);
    }
#endif
    __syncthreads();

    float m_run[8], l_run[8];
    #pragma unroll
    for (int v = 0; v < 8; ++v) { m_run[v] = -1e30f; l_run[v] = 0.0f; }
    v8f oacc[8] = {};                    // 16 rows x 128 d_v cols

    const float scale = 0.044194173824159216f;   // 1/sqrt(512)

    for (int ktile = 0; ktile < SS; ktile += 64) {
        const int key = ktile + dw * 16 + ln;
        const unsigned short* krow = Kb + ((long)b * SS + key) * DD;

        if (ktile + 64 < SS) {           // prefetch next key tile rows
            __builtin_prefetch(krow + (long)64 * DD, 0, 0);
            __builtin_prefetch(Vt + ((long)b * DD + dw * 128 + ln) * SS + ktile + 64, 0, 0);
        }

        // ---- scores: 16x16 frag, K-loop over D=512 ----
        v8f sc = {};
        for (int kt = 0; kt < DD; kt += 32) {
            union { v16bf v; uint4 q[2]; } A, Bf;
            const unsigned short* aptr = &sQ[(qw * 16 + ln) * QSTR + kt + half * 8];
            A.q[0] = *(const uint4*)(aptr);
            A.q[1] = *(const uint4*)(aptr + 16);
            const uint4* bq = (const uint4*)(krow + kt + half * 8);
            Bf.q[0] = bq[0]; Bf.q[1] = bq[2];
            sc = __builtin_amdgcn_wmma_f32_16x16x32_bf16(
                false, A.v, false, Bf.v, (short)0, sc, false, false);
        }

        // ---- scale + per-row max within fragment (16-lane butterfly) ----
        float rmax[8];
        #pragma unroll
        for (int v = 0; v < 8; ++v) {
            float s = sc[v] * scale;
            sc[v] = s;
            float mx = s;
            #pragma unroll
            for (int off = 1; off < 16; off <<= 1)
                mx = fmaxf(mx, __shfl_xor(mx, off, 32));
            rmax[v] = mx;
        }
        if (ln == 0) {
            #pragma unroll
            for (int v = 0; v < 8; ++v) sMax[qw][dw][8 * half + v] = rmax[v];
        }
        __syncthreads();

        // ---- combine max across the 4 key sub-tiles, running max update ----
        float mnew[8], corr[8];
        #pragma unroll
        for (int v = 0; v < 8; ++v) {
            int r = 8 * half + v;
            float mt = fmaxf(fmaxf(sMax[qw][0][r], sMax[qw][1][r]),
                             fmaxf(sMax[qw][2][r], sMax[qw][3][r]));
            mnew[v] = fmaxf(m_run[v], mt);
            corr[v] = __expf(m_run[v] - mnew[v]);
            m_run[v] = mnew[v];
        }

        // ---- p = exp(s - m), row partial sums, publish P tile as bf16 ----
        float rsum[8];
        #pragma unroll
        for (int v = 0; v < 8; ++v) {
            float p = __expf(sc[v] - mnew[v]);
            sc[v] = p;
            float sm = p;
            #pragma unroll
            for (int off = 1; off < 16; off <<= 1)
                sm += __shfl_xor(sm, off, 32);
            rsum[v] = sm;
        }
        if (ln == 0) {
            #pragma unroll
            for (int v = 0; v < 8; ++v) sSum[qw][dw][8 * half + v] = rsum[v];
        }
        #pragma unroll
        for (int v = 0; v < 8; ++v)
            sP[qw][v + 8 * half][dw * 16 + ln] = f2bf(sc[v]);
        __syncthreads();

        // ---- l update + rescale output accumulators ----
        #pragma unroll
        for (int v = 0; v < 8; ++v) {
            int r = 8 * half + v;
            float st = sSum[qw][0][r] + sSum[qw][1][r] + sSum[qw][2][r] + sSum[qw][3][r];
            l_run[v] = l_run[v] * corr[v] + st;
        }
        #pragma unroll
        for (int t = 0; t < 8; ++t)
            #pragma unroll
            for (int v = 0; v < 8; ++v)
                oacc[t][v] *= corr[v];

        // ---- oacc += P(16x64) @ V(64 x 128-slice) ----
        #pragma unroll
        for (int kk = 0; kk < 2; ++kk) {
            union { v16bf v; uint4 q[2]; } Ap;
            const unsigned short* pp = &sP[qw][ln][kk * 32 + half * 8];
            Ap.q[0] = *(const uint4*)pp;
            Ap.q[1] = *(const uint4*)(pp + 16);
            #pragma unroll
            for (int t = 0; t < 8; ++t) {
                int n = dw * 128 + t * 16 + ln;     // d_v column
                const uint4* vp = (const uint4*)(Vt + ((long)b * DD + n) * SS
                                                 + ktile + kk * 32 + half * 8);
                union { v16bf v; uint4 q[2]; } Bv;
                Bv.q[0] = vp[0]; Bv.q[1] = vp[2];
                oacc[t] = __builtin_amdgcn_wmma_f32_16x16x32_bf16(
                    false, Ap.v, false, Bv.v, (short)0, oacc[t], false, false);
            }
        }
    }

    // ---- epilogue: divide by l, store f32 ----
    #pragma unroll
    for (int v = 0; v < 8; ++v) {
        float inv = 1.0f / l_run[v];
        int s = q0 + qw * 16 + v + 8 * half;
        float* op = Out + ((long)b * SS + s) * DD + dw * 128;
        #pragma unroll
        for (int t = 0; t < 8; ++t)
            op[t * 16 + ln] = oacc[t][v] * inv;
    }
}

// ---------------------------------------------------------------------------
// Launcher
// ---------------------------------------------------------------------------
extern "C" void kernel_launch(void* const* d_in, const int* in_sizes, int n_in,
                              void* d_out, int out_size, void* d_ws, size_t ws_size,
                              hipStream_t stream) {
    const float* x_q = (const float*)d_in[0];
    const float* x_k = (const float*)d_in[1];
    const float* x_v = (const float*)d_in[2];
    const float* Wq  = (const float*)d_in[3];
    const float* bq  = (const float*)d_in[4];
    const float* Wk  = (const float*)d_in[5];
    const float* bk  = (const float*)d_in[6];
    const float* Wv  = (const float*)d_in[7];
    const float* bv  = (const float*)d_in[8];
    float* out = (float*)d_out;

    char* ws = (char*)d_ws;
    const size_t QKV = (size_t)BB * SS * DD * sizeof(unsigned short); // 32 MiB each
    unsigned short* Qb  = (unsigned short*)(ws);
    unsigned short* Kb  = (unsigned short*)(ws + QKV);
    unsigned short* VtB = (unsigned short*)(ws + 2 * QKV);
    unsigned short* Wtq = (unsigned short*)(ws + 3 * QKV);
    unsigned short* Wtk = Wtq + (size_t)DD * FF;
    unsigned short* Wtv = Wtk + (size_t)DD * FF;

    dim3 wtb(256), wtg((DD * FF) / 256);
    wt_kernel<<<wtg, wtb, 0, stream>>>(Wq, Wtq);
    wt_kernel<<<wtg, wtb, 0, stream>>>(Wk, Wtk);
    wt_kernel<<<wtg, wtb, 0, stream>>>(Wv, Wtv);

    dim3 pb(256), pg((BB * SS) / 128, DD / 64);
    proj_kernel<<<pg, pb, 0, stream>>>(x_q, Wtq, bq, Qb,  0);
    proj_kernel<<<pg, pb, 0, stream>>>(x_k, Wtk, bk, Kb,  0);
    proj_kernel<<<pg, pb, 0, stream>>>(x_v, Wtv, bv, VtB, 1);

    dim3 ab(256), ag(SS / 32, BB);
    attn_kernel<<<ag, ab, 0, stream>>>(Qb, Kb, VtB, out);
}
